// CgpHmmLayer_4389456577425
// MI455X (gfx1250) — matrix operand order repaired and checked by reading, not compiled, not run on hardware
//
#include <hip/hip_runtime.h>
#include <cstddef>
#include <cstdint>

typedef unsigned short ushort_t;
typedef unsigned int   uint_t;

typedef __attribute__((ext_vector_type(16))) __bf16 v16bf;
typedef __attribute__((ext_vector_type(8)))  float  v8f;

#define BATCH 32
#define TT    4096
#define SS    308
#define SP    312            // padded A-column length (16B-aligned: 312*2 = 624 B)
#define EE    126
#define TC    512
#define NCHUNK (TT / TC)

// ---------------- helpers ----------------

__device__ __forceinline__ ushort_t f2bf(float f) {
    // round-to-nearest-even f32 -> bf16 (inputs are probabilities; no NaN path needed)
    uint_t u = __float_as_uint(f);
    uint_t r = u + 0x7FFFu + ((u >> 16) & 1u);
    return (ushort_t)(r >> 16);
}

union BF16x16 { uint4 q[2]; v16bf v; };

// acc += (bf16)src0.lo16 * f32   |   acc += (bf16)src0.hi16 * f32
// V_FMA_MIX_F32_BF16: {op_sel_hi[i],op_sel[i]} = 2 -> src[15:0] bf16, 3 -> src[31:16] bf16
#define FMA_MIX_LO(acc, u, a)                                                  \
    asm("v_fma_mix_f32_bf16 %0, %1, %2, %0 op_sel_hi:[1,0,0]"                  \
        : "+v"(acc) : "v"(u), "v"(a))
#define FMA_MIX_HI(acc, u, a)                                                  \
    asm("v_fma_mix_f32_bf16 %0, %1, %2, %0 op_sel:[1,0,0] op_sel_hi:[1,0,0]"   \
        : "+v"(acc) : "v"(u), "v"(a))

// ---------------- kernel 0: A -> bf16, column-major, padded ----------------
// Acm[j*SP + k] = bf16(A[k*SS + j]) for k < SS, else 0. One thread per element i = j*SP + k.

__global__ void conv_a_kernel(const float* __restrict__ A, ushort_t* __restrict__ Acm) {
    int i = blockIdx.x * blockDim.x + threadIdx.x;
    if (i < SS * SP) {
        int j = i / SP;
        int k = i - j * SP;
        Acm[i] = (k < SS) ? f2bf(A[k * SS + j]) : (ushort_t)0;
    }
}

// ---------------- kernel 1: emission GEMM chunk (WMMA bf16) ----------------
// emis[b][tc][s] = sum_e inputs[b][t0+tc][e] * Bm[s][e], for tc in [0,TC).
// Grid: 256 WGs x 256 threads. Each WG: 64 chunk-rows (4 M-tiles) x 20 N-tiles, K padded 126->128.

#define P1_LDS (64 * 128 * 2 + 320 * 128 * 2)  // inTile + Btile (bf16) = 98304 B

__global__ __launch_bounds__(256) void emis_gemm_kernel(const float* __restrict__ inp,
                                                        const float* __restrict__ Bm,
                                                        float* __restrict__ emis,
                                                        int t0) {
    extern __shared__ char smem[];
    ushort_t* inT = (ushort_t*)smem;                  // [64][128] bf16
    ushort_t* Bt  = (ushort_t*)(smem + 64 * 128 * 2); // [320][128] bf16

    const int tid     = threadIdx.x;
    const int rowBase = blockIdx.x * 64;          // chunk row = b*TC + tc
    const int b       = rowBase >> 9;             // TC = 512
    const int tcB     = rowBase & (TC - 1);
    const float* inRow0 = inp + ((size_t)b * TT + (size_t)t0 + (size_t)tcB) * EE;

    // stage inputs tile (f32 -> bf16, zero-pad K to 128)
    for (int i = tid; i < 64 * 128; i += 256) {
        int r = i >> 7, e = i & 127;
        float v = (e < EE) ? inRow0[(size_t)r * EE + e] : 0.f;
        inT[i] = f2bf(v);
    }
    // stage full B matrix (f32 -> bf16, zero-pad S to 320 and K to 128)
    for (int i = tid; i < 320 * 128; i += 256) {
        int s = i >> 7, e = i & 127;
        float v = (s < SS && e < EE) ? Bm[s * EE + e] : 0.f;
        Bt[i] = f2bf(v);
    }
    __syncthreads();

    const int lane  = tid & 31;
    const int wv    = tid >> 5;
    const int mt    = wv >> 1;           // 0..3 : M-tile
    const int nb    = (wv & 1) * 10;     // N-tile range base
    const int mrow  = mt * 16 + (lane & 15);
    const int khalf = lane >> 4;         // 0/1

    // A-operand fragments (16x32 bf16): halves 0-7 -> K = kc*32 + khalf*8 + h,
    //                                   halves 8-15 -> K = kc*32 + 16 + khalf*8 + (h-8)
    BF16x16 afr[4];
#pragma unroll
    for (int kc = 0; kc < 4; ++kc) {
        afr[kc].q[0] = *(const uint4*)&inT[mrow * 128 + kc * 32 + khalf * 8];
        afr[kc].q[1] = *(const uint4*)&inT[mrow * 128 + kc * 32 + 16 + khalf * 8];
    }

    for (int nt = 0; nt < 10; ++nt) {
        const int n0   = (nb + nt) * 16;
        const int srow = n0 + (lane & 15);   // output column s; also B row in [s][e] storage
        v8f acc = {0.f, 0.f, 0.f, 0.f, 0.f, 0.f, 0.f, 0.f};
#pragma unroll
        for (int kc = 0; kc < 4; ++kc) {
            // B-operand (32x16 bf16): lane holds column n = lane&15,
            // halves 0-7 -> K = kc*32 + khalf*8 + h, halves 8-15 -> K = kc*32 + 16 + khalf*8 + (h-8)
            BF16x16 bfr;
            const uint4* q = (const uint4*)&Bt[srow * 128 + kc * 32 + khalf * 8];
            bfr.q[0] = q[0];
            bfr.q[1] = q[2];
            acc = __builtin_amdgcn_wmma_f32_16x16x32_bf16(
                false, afr[kc].v, false, bfr.v, (short)0, acc, false, false);
        }
        if (srow < SS) {
            // D layout: VGPR vr, lanes 0-15 -> M = vr, lanes 16-31 -> M = vr + 8
            const int gr = rowBase + mt * 16 + 8 * khalf;
#pragma unroll
            for (int vr = 0; vr < 8; ++vr)
                emis[(size_t)(gr + vr) * SS + srow] = acc[vr];
        }
    }
}

// ---------------- kernel 2: HMM forward scan chunk ----------------
// One workgroup per batch (32 WGs x 320 threads = 10 waves).
// A (bf16, column-major, padded to SP) resident in LDS; alpha double-buffered in LDS.
// Inner dot product: ds_load_b128 A-column + float4 alpha broadcasts + v_fma_mix_f32_bf16.

#define P2_LDS (SS * SP * 2 + 2 * 320 * 4 + 16 * 4 + 16)  // 194832 B

__device__ __forceinline__ float block_reduce_320(float v, float* red, float* zsh) {
#pragma unroll
    for (int off = 16; off > 0; off >>= 1) v += __shfl_xor(v, off, 32);
    const int lane = threadIdx.x & 31;
    const int w    = threadIdx.x >> 5;
    if (lane == 0) red[w] = v;
    __syncthreads();
    if (threadIdx.x == 0) {
        float z = 0.f;
#pragma unroll
        for (int i = 0; i < 10; ++i) z += red[i];
        *zsh = z;
    }
    __syncthreads();
    return *zsh;
}

__global__ __launch_bounds__(320) void hmm_scan_kernel(const float* __restrict__ emis,
                                                       const ushort_t* __restrict__ Acm,
                                                       const float* __restrict__ Ivec,
                                                       float* __restrict__ alphaSt,
                                                       float* __restrict__ llSt,
                                                       float* __restrict__ out,
                                                       int first, int last) {
    extern __shared__ char smem[];
    ushort_t* ldsA  = (ushort_t*)smem;                       // [SS columns][SP] bf16
    float* alpha0   = (float*)(smem + SS * SP * 2);          // 320 floats (16B aligned)
    float* alpha1   = alpha0 + 320;
    float* red      = alpha1 + 320;                          // 16 floats
    float* zsh      = red + 16;

    const int b   = blockIdx.x;
    const int tid = threadIdx.x;
    const int j   = tid;
    const bool act = (j < SS);

    // cooperative load of A into LDS (uint4 granularity: SS*SP halves = 12012 uint4)
    {
        const uint4* src = (const uint4*)Acm;
        uint4* dst = (uint4*)ldsA;
        for (int i = tid; i < (SS * SP) / 8; i += 320) dst[i] = src[i];
    }
    // zero alpha pads (indices 308..319) so float4 reads are harmless
    if (tid >= SS) { alpha0[tid] = 0.f; alpha1[tid] = 0.f; }

    float* cur = alpha0;
    float* nxt = alpha1;
    float ll = 0.f;
    int tcStart;

    if (first) {
        float a = 0.f;
        if (act) a = Ivec[j] * emis[(size_t)(b * TC + 0) * SS + j];
        __syncthreads();  // ldsA + pad zeroing visible
        float z = block_reduce_320(a, red, zsh);
        ll = logf(z);
        if (act) cur[j] = a / z;
        tcStart = 1;
    } else {
        if (act) cur[j] = alphaSt[b * SS + j];
        tcStart = 0;
    }
    __syncthreads();

    for (int tc = tcStart; tc < TC; ++tc) {
        const float* erow = &emis[(size_t)(b * TC + tc) * SS];
        if (tc + 1 < TC && act && ((j & 15) == 0))
            __builtin_prefetch(&emis[(size_t)(b * TC + tc + 1) * SS + j], 0, 0);

        float acc0 = 0.f, acc1 = 0.f, acc2 = 0.f, acc3 = 0.f;
        if (act) {
            const uint4*  colA = (const uint4*)&ldsA[j * SP];  // 39 uint4 = 312 bf16
            const float4* al   = (const float4*)cur;           // broadcast reads
#pragma unroll 3
            for (int q = 0; q < SP / 8; ++q) {
                uint4  ua = colA[q];
                float4 a0 = al[2 * q];
                float4 a1 = al[2 * q + 1];
                FMA_MIX_LO(acc0, ua.x, a0.x);
                FMA_MIX_HI(acc1, ua.x, a0.y);
                FMA_MIX_LO(acc2, ua.y, a0.z);
                FMA_MIX_HI(acc3, ua.y, a0.w);
                FMA_MIX_LO(acc0, ua.z, a1.x);
                FMA_MIX_HI(acc1, ua.z, a1.y);
                FMA_MIX_LO(acc2, ua.w, a1.z);
                FMA_MIX_HI(acc3, ua.w, a1.w);
            }
        }
        float anew = act ? ((acc0 + acc1) + (acc2 + acc3)) * erow[j] : 0.f;

        float z = block_reduce_320(anew, red, zsh);
        ll += logf(z);
        if (act) nxt[j] = anew / z;
        __syncthreads();
        float* t = cur; cur = nxt; nxt = t;
    }

    if (act) alphaSt[b * SS + j] = cur[j];
    if (tid == 0) {
        float tot = (first ? 0.f : llSt[b]) + ll;
        llSt[b] = tot;
        if (last) out[b] = tot;
    }
}

// ---------------- host launcher ----------------

extern "C" void kernel_launch(void* const* d_in, const int* in_sizes, int n_in,
                              void* d_out, int out_size, void* d_ws, size_t ws_size,
                              hipStream_t stream) {
    (void)in_sizes; (void)n_in; (void)out_size; (void)ws_size;

    const float* inp = (const float*)d_in[0];  // [32,4096,126]
    const float* A   = (const float*)d_in[1];  // [308,308]
    const float* Bm  = (const float*)d_in[2];  // [308,126]
    const float* Iv  = (const float*)d_in[3];  // [308]
    float* out = (float*)d_out;                // [32]

    char* ws = (char*)d_ws;
    size_t off = 0;
    float* emis = (float*)(ws + off);                 // 32*512*308*4 = 20,185,088 B
    off += (size_t)BATCH * TC * SS * sizeof(float);
    off = (off + 255) & ~(size_t)255;
    ushort_t* Acm = (ushort_t*)(ws + off);            // 308*312*2 = 192,192 B
    off += (size_t)SS * SP * sizeof(ushort_t);
    off = (off + 255) & ~(size_t)255;
    float* alphaSt = (float*)(ws + off);              // 32*308*4
    off += (size_t)BATCH * SS * sizeof(float);
    off = (off + 255) & ~(size_t)255;
    float* llSt = (float*)(ws + off);                 // 32*4

    // allow >64KB dynamic LDS (WGP-class allocations on CDNA5); deterministic per call
    (void)hipFuncSetAttribute((const void*)emis_gemm_kernel,
                              hipFuncAttributeMaxDynamicSharedMemorySize, P1_LDS);
    (void)hipFuncSetAttribute((const void*)hmm_scan_kernel,
                              hipFuncAttributeMaxDynamicSharedMemorySize, P2_LDS);

    conv_a_kernel<<<(SS * SP + 255) / 256, 256, 0, stream>>>(A, Acm);

    for (int c = 0; c < NCHUNK; ++c) {
        emis_gemm_kernel<<<(BATCH * TC) / 64, 256, P1_LDS, stream>>>(inp, Bm, emis, c * TC);
        hmm_scan_kernel<<<BATCH, 320, P2_LDS, stream>>>(emis, Acm, Iv, alphaSt, llSt, out,
                                                        (c == 0) ? 1 : 0,
                                                        (c == NCHUNK - 1) ? 1 : 0);
    }
}
